// EdgeDegreeEmbeddingNetwork_16183436771998
// MI455X (gfx1250) — matrix operand order repaired and checked by reading, not compiled, not run on hardware
//
#include <hip/hip_runtime.h>

// ---------------------------------------------------------------------------
// EdgeDegreeEmbedding for MI455X (gfx1250, wave32, WMMA)
//
// Math reduction (see analysis): xs[1], xs[2] are zero; xs[0] is a constant
// vector x0 = exp_w + exp_b for every edge.  Everything collapses to:
//   h1 = silu(LN(edge_scalars @ W1 + b1))          (E,128)@(128,64)
//   h2 = silu(LN(h1 @ W2 + b2))                    (E,64)@(64,64)
//   Q  = h2 @ B + c                                (E,64)@(64,224)
//   p0[o]   = s0*(Q[:,o]+c) + proj_b ; p1[o,k] = y1[k]*(Q[:,128+o]+c) ; p2 ~5
//   out[dst] += concat(p0,p1,p2)/sqrt(8)           (atomic scatter, L2-resident)
// B = rad_w3[:,0:384-blocks] * diag(x0) * proj_w{0,1,2}[0:128,:]  (64x224)
// ---------------------------------------------------------------------------

typedef _Float16 v16h __attribute__((ext_vector_type(16)));
typedef _Float16 v8h  __attribute__((ext_vector_type(8)));
typedef float    v8f  __attribute__((ext_vector_type(8)));

#define N_BASIS   128
#define FC_HID    64
#define QCOLS     224   // 128 (l=0) + 64 (l=1) + 32 (l=2)
#define OUT_DIM   480   // 128*1 + 64*3 + 32*5
#define TILE_E    128   // edges per block (8 waves x 16)

// Load a 16-half WMMA fragment from two 16-byte-aligned 8-half chunks.
__device__ __forceinline__ v16h ld16(const _Float16* p0, const _Float16* p1) {
  v8h a = *(const v8h*)p0;
  v8h b = *(const v8h*)p1;
  v16h r;
#pragma unroll
  for (int i = 0; i < 8; ++i) { r[i] = a[i]; r[i + 8] = b[i]; }
  return r;
}

__device__ __forceinline__ float silu(float x) {
  return x / (1.0f + __expf(-x));
}

// ---------------------------------------------------------------------------
// Pre-pass kernels (cheap, run each launch; graph-capture safe)
// ---------------------------------------------------------------------------
__global__ void k_cvt_f16(const float* __restrict__ src,
                          _Float16* __restrict__ dst, int n) {
  for (int i = blockIdx.x * blockDim.x + threadIdx.x; i < n;
       i += gridDim.x * blockDim.x)
    dst[i] = (_Float16)src[i];
}

// w is (K x N) row-major f32 -> wt is (N x K) row-major f16
__global__ void k_transpose_f16(const float* __restrict__ w,
                                _Float16* __restrict__ wt, int K, int N) {
  int i = blockIdx.x * blockDim.x + threadIdx.x;
  if (i >= K * N) return;
  int k = i / N, n = i % N;
  wt[n * K + k] = (_Float16)w[i];
}

// Build BT (224 x 64, f16, transposed for WMMA-B loads) and c (224, f32).
__global__ void k_build_bt(const float* __restrict__ rad_w3,    // (64,960)
                           const float* __restrict__ rad_off,   // (960)
                           const float* __restrict__ exp_w,
                           const float* __restrict__ exp_b,
                           const float* __restrict__ pw0,       // (224,128)
                           const float* __restrict__ pw1,       // (384,64)
                           const float* __restrict__ pw2,       // (352,32)
                           _Float16* __restrict__ BT,           // (224,64)
                           float* __restrict__ cvec) {          // (224)
  int i = blockIdx.x * blockDim.x + threadIdx.x;     // o*64 + h
  if (i >= QCOLS * FC_HID) return;
  int o = i >> 6, h = i & 63;
  int cb, ow, ps;
  const float* P;
  if (o < 128)      { cb = 0;   P = pw0; ps = 128; ow = o; }
  else if (o < 192) { cb = 128; P = pw1; ps = 64;  ow = o - 128; }
  else              { cb = 256; P = pw2; ps = 32;  ow = o - 192; }
  float acc = 0.0f;
  for (int u = 0; u < 128; ++u) {
    float x0 = exp_w[u] + exp_b[u];
    acc += rad_w3[h * 960 + cb + u] * x0 * P[u * ps + ow];
  }
  BT[o * FC_HID + h] = (_Float16)acc;
  if (h == 0) {
    float c = 0.0f;
    for (int u = 0; u < 128; ++u)
      c += rad_off[cb + u] * (exp_w[u] + exp_b[u]) * P[u * ps + ow];
    cvec[o] = c;
  }
}

__global__ void k_zero(float* __restrict__ p, int n) {
  for (int i = blockIdx.x * blockDim.x + threadIdx.x; i < n;
       i += gridDim.x * blockDim.x)
    p[i] = 0.0f;
}

// ---------------------------------------------------------------------------
// Main fused edge kernel: 3 WMMA GEMM stages + 2 LayerNorm/SiLU + scatter
// ---------------------------------------------------------------------------
__global__ __launch_bounds__(256, 1)
void k_edge(const float* __restrict__ edge_attr,   // (E,9)
            const int*   __restrict__ edge_dst,    // (E)
            const _Float16* __restrict__ Xh,       // (E,128) f16
            const _Float16* __restrict__ W1T,      // (64,128) f16
            const _Float16* __restrict__ W2T,      // (64,64)  f16
            const _Float16* __restrict__ BT,       // (224,64) f16
            const float* __restrict__ cvec,        // (224)
            const float* __restrict__ b1,  const float* __restrict__ g1,
            const float* __restrict__ bl1, const float* __restrict__ b2,
            const float* __restrict__ g2,  const float* __restrict__ bl2,
            const float* __restrict__ proj_b,      // (128)
            float* __restrict__ out, int E) {
  // per-wave private staging (no cross-wave sharing -> no barriers needed)
  __shared__ __align__(16) _Float16 Hs[8][16][72];   // 72-half row stride (16B-mult)

  const int tid = threadIdx.x;
  const int w   = tid >> 5;        // wave id 0..7 (wave32)
  const int l   = tid & 31;        // lane
  const int ln  = l & 15;          // lane-in-half
  const int hi  = l >> 4;          // half-wave id
  const int e0  = blockIdx.x * TILE_E + w * 16;   // first edge of wave tile

  // ---- GEMM1: (16x128) @ (128x64) -> acc1[4] (16x16 f32 tiles) -------------
  int ea = min(e0 + ln, E - 1);
  const _Float16* arow = Xh + (size_t)ea * N_BASIS;

  v8f acc1[4];
#pragma unroll
  for (int nt = 0; nt < 4; ++nt) acc1[nt] = (v8f)(0.0f);

#pragma unroll
  for (int kc = 0; kc < 4; ++kc) {
    const int kb = kc * 32 + hi * 8;
    v16h a = ld16(arow + kb, arow + kb + 16);
#pragma unroll
    for (int nt = 0; nt < 4; ++nt) {
      const _Float16* bp = W1T + (16 * nt + ln) * N_BASIS + kc * 32 + hi * 8;
      v16h b = ld16(bp, bp + 16);
      acc1[nt] = __builtin_amdgcn_wmma_f32_16x16x32_f16(
          false, a, false, b, (short)0, acc1[nt], false, false);
    }
  }

  // per-lane LN/affine params for columns n = 16*nt + ln
  float b1v[4], g1v[4], bl1v[4], b2v[4], g2v[4], bl2v[4];
#pragma unroll
  for (int nt = 0; nt < 4; ++nt) {
    int n = 16 * nt + ln;
    b1v[nt] = b1[n]; g1v[nt] = g1[n]; bl1v[nt] = bl1[n];
    b2v[nt] = b2[n]; g2v[nt] = g2[n]; bl2v[nt] = bl2[n];
  }

  // ---- LN1 + SiLU, restage as f16 in LDS (rows of H1 live in a half-wave) --
#pragma unroll
  for (int vr = 0; vr < 8; ++vr) {
    float s = 0.0f, ss = 0.0f;
#pragma unroll
    for (int nt = 0; nt < 4; ++nt) {
      float v = acc1[nt][vr] + b1v[nt];
      s += v; ss += v * v;
    }
#pragma unroll
    for (int off = 1; off <= 8; off <<= 1) {   // 16-lane half-wave reduction
      s  += __shfl_xor(s,  off);
      ss += __shfl_xor(ss, off);
    }
    float mu = s * (1.0f / 64.0f);
    float rs = rsqrtf(ss * (1.0f / 64.0f) - mu * mu + 1e-5f);
    int m = vr + 8 * hi;
#pragma unroll
    for (int nt = 0; nt < 4; ++nt) {
      float x = (acc1[nt][vr] + b1v[nt] - mu) * rs * g1v[nt] + bl1v[nt];
      Hs[w][m][16 * nt + ln] = (_Float16)silu(x);
    }
  }

  // ---- GEMM2: (16x64) @ (64x64) -> acc2[4] ---------------------------------
  const _Float16* hrow = &Hs[w][ln][0];
  v8f acc2[4];
#pragma unroll
  for (int nt = 0; nt < 4; ++nt) acc2[nt] = (v8f)(0.0f);

#pragma unroll
  for (int kc = 0; kc < 2; ++kc) {
    const int kb = kc * 32 + hi * 8;
    v16h a = ld16(hrow + kb, hrow + kb + 16);
#pragma unroll
    for (int nt = 0; nt < 4; ++nt) {
      const _Float16* bp = W2T + (16 * nt + ln) * FC_HID + kc * 32 + hi * 8;
      v16h b = ld16(bp, bp + 16);
      acc2[nt] = __builtin_amdgcn_wmma_f32_16x16x32_f16(
          false, a, false, b, (short)0, acc2[nt], false, false);
    }
  }

  // ---- LN2 + SiLU, restage --------------------------------------------------
#pragma unroll
  for (int vr = 0; vr < 8; ++vr) {
    float s = 0.0f, ss = 0.0f;
#pragma unroll
    for (int nt = 0; nt < 4; ++nt) {
      float v = acc2[nt][vr] + b2v[nt];
      s += v; ss += v * v;
    }
#pragma unroll
    for (int off = 1; off <= 8; off <<= 1) {
      s  += __shfl_xor(s,  off);
      ss += __shfl_xor(ss, off);
    }
    float mu = s * (1.0f / 64.0f);
    float rs = rsqrtf(ss * (1.0f / 64.0f) - mu * mu + 1e-5f);
    int m = vr + 8 * hi;
#pragma unroll
    for (int nt = 0; nt < 4; ++nt) {
      float x = (acc2[nt][vr] + b2v[nt] - mu) * rs * g2v[nt] + bl2v[nt];
      Hs[w][m][16 * nt + ln] = (_Float16)silu(x);
    }
  }

  // ---- GEMM3: (16x64) @ (64x224) -> acc3[14] -------------------------------
  v8f acc3[14];
#pragma unroll
  for (int nt = 0; nt < 14; ++nt) acc3[nt] = (v8f)(0.0f);

#pragma unroll
  for (int kc = 0; kc < 2; ++kc) {
    const int kb = kc * 32 + hi * 8;
    v16h a = ld16(hrow + kb, hrow + kb + 16);
#pragma unroll
    for (int nt = 0; nt < 14; ++nt) {
      const _Float16* bp = BT + (16 * nt + ln) * FC_HID + kc * 32 + hi * 8;
      v16h b = ld16(bp, bp + 16);
      acc3[nt] = __builtin_amdgcn_wmma_f32_16x16x32_f16(
          false, a, false, b, (short)0, acc3[nt], false, false);
    }
  }

  // ---- Epilogue: Q + c, per-l tensor-product expansion, atomic scatter -----
  float cn[14];
#pragma unroll
  for (int nt = 0; nt < 14; ++nt) cn[nt] = cvec[16 * nt + ln];
  float pbv[8];
#pragma unroll
  for (int nt = 0; nt < 8; ++nt) pbv[nt] = proj_b[16 * nt + ln];

  const float invs = 0.35355339059327373f;   // 1/sqrt(AVG_AGG=8)

#pragma unroll
  for (int vr = 0; vr < 8; ++vr) {
    int e = e0 + vr + 8 * hi;                // D-layout row -> edge index
    if (e >= E) continue;
    int dst = edge_dst[e];
    float* ob = out + (size_t)dst * OUT_DIM;
    const float* at = edge_attr + (size_t)e * 9;
    float s0 = at[0];
    float y1a = at[1], y1b = at[2], y1c = at[3];
    float y2a = at[4], y2b = at[5], y2c = at[6], y2d = at[7], y2e = at[8];
#pragma unroll
    for (int nt = 0; nt < 14; ++nt) {
      int n = 16 * nt + ln;
      float q = acc3[nt][vr] + cn[nt];
      if (nt < 8) {                                    // l=0 block (n<128)
        atomicAdd(ob + n, (q * s0 + pbv[nt]) * invs);
      } else if (nt < 12) {                            // l=1 block
        int o = n - 128;
        atomicAdd(ob + 128 + o * 3 + 0, q * y1a * invs);
        atomicAdd(ob + 128 + o * 3 + 1, q * y1b * invs);
        atomicAdd(ob + 128 + o * 3 + 2, q * y1c * invs);
      } else {                                         // l=2 block
        int o = n - 192;
        atomicAdd(ob + 320 + o * 5 + 0, q * y2a * invs);
        atomicAdd(ob + 320 + o * 5 + 1, q * y2b * invs);
        atomicAdd(ob + 320 + o * 5 + 2, q * y2c * invs);
        atomicAdd(ob + 320 + o * 5 + 3, q * y2d * invs);
        atomicAdd(ob + 320 + o * 5 + 4, q * y2e * invs);
      }
    }
  }
}

// ---------------------------------------------------------------------------
extern "C" void kernel_launch(void* const* d_in, const int* in_sizes, int n_in,
                              void* d_out, int out_size, void* d_ws,
                              size_t ws_size, hipStream_t stream) {
  // setup_inputs() order:
  const float* edge_attr    = (const float*)d_in[1];
  const float* edge_scalars = (const float*)d_in[2];
  const int*   edge_dst     = (const int*)d_in[4];
  const float* exp_w        = (const float*)d_in[6];
  const float* exp_b        = (const float*)d_in[7];
  const float* rad_w1       = (const float*)d_in[8];
  const float* rad_b1       = (const float*)d_in[9];
  const float* rad_ln1_g    = (const float*)d_in[10];
  const float* rad_ln1_b    = (const float*)d_in[11];
  const float* rad_w2       = (const float*)d_in[12];
  const float* rad_b2       = (const float*)d_in[13];
  const float* rad_ln2_g    = (const float*)d_in[14];
  const float* rad_ln2_b    = (const float*)d_in[15];
  const float* rad_w3       = (const float*)d_in[16];
  const float* rad_offset   = (const float*)d_in[17];
  const float* proj_w0      = (const float*)d_in[18];
  const float* proj_w1      = (const float*)d_in[19];
  const float* proj_w2      = (const float*)d_in[20];
  const float* proj_b       = (const float*)d_in[21];

  const int E = in_sizes[1] / 9;

  // workspace layout (f16 staging + fused B/c), ~20.6 MB
  char* ws = (char*)d_ws;
  _Float16* Xh  = (_Float16*)ws;                 size_t off = (size_t)E * N_BASIS * 2;
  _Float16* W1T = (_Float16*)(ws + off);         off += (size_t)FC_HID * N_BASIS * 2;
  _Float16* W2T = (_Float16*)(ws + off);         off += (size_t)FC_HID * FC_HID * 2;
  _Float16* BTh = (_Float16*)(ws + off);         off += (size_t)QCOLS * FC_HID * 2;
  float*    cv  = (float*)(ws + off);

  // pre-passes
  k_cvt_f16<<<4096, 256, 0, stream>>>(edge_scalars, Xh, E * N_BASIS);
  k_transpose_f16<<<(N_BASIS * FC_HID + 255) / 256, 256, 0, stream>>>(
      rad_w1, W1T, N_BASIS, FC_HID);
  k_transpose_f16<<<(FC_HID * FC_HID + 255) / 256, 256, 0, stream>>>(
      rad_w2, W2T, FC_HID, FC_HID);
  k_build_bt<<<(QCOLS * FC_HID + 255) / 256, 256, 0, stream>>>(
      rad_w3, rad_offset, exp_w, exp_b, proj_w0, proj_w1, proj_w2, BTh, cv);
  k_zero<<<2048, 256, 0, stream>>>((float*)d_out, out_size);

  // main fused kernel: 8 waves/block, 16 edges/wave
  k_edge<<<(E + TILE_E - 1) / TILE_E, 256, 0, stream>>>(
      edge_attr, edge_dst, Xh, W1T, W2T, BTh, cv,
      rad_b1, rad_ln1_g, rad_ln1_b, rad_b2, rad_ln2_g, rad_ln2_b,
      proj_b, (float*)d_out, E);
}
